// TransformerDecoderLayer_56770877718567
// MI455X (gfx1250) — compile-verified, matrix operand order checked by reading, not run on hardware
//
#include <hip/hip_runtime.h>
#include <math.h>

typedef __bf16 v16bf __attribute__((ext_vector_type(16)));
typedef __bf16 v8bf  __attribute__((ext_vector_type(8)));
typedef float  v8f   __attribute__((ext_vector_type(8)));

#define D_MODEL 1024
#define N_HEAD  16
#define DH      64

__device__ __forceinline__ v8f wmma_bf16(v16bf a, v16bf b, v8f c) {
  // v_wmma_f32_16x16x32_bf16  D = A*B + C   (fp32 accumulate)
  return __builtin_amdgcn_wmma_f32_16x16x32_bf16(false, a, false, b, (short)0, c,
                                                 false, false);
}

// A/B fragment, 16-bit 16x32 layout: lane holds row (lane&15), K chunks at
// kb and kb+16 where kb = (lane>>4)*8 -> two contiguous 16B LDS loads.
__device__ __forceinline__ v16bf load_frag16(const __bf16* p) {
  v8bf lo = *(const v8bf*)(p);
  v8bf hi = *(const v8bf*)(p + 16);
  return __builtin_shufflevector(lo, hi, 0,1,2,3,4,5,6,7,8,9,10,11,12,13,14,15);
}

__device__ __forceinline__ float gelu_exact(float x) {
  return 0.5f * x * (1.0f + erff(x * 0.70710678118654752f));
}

// ---------------------------------------------------------------------------
// Generic GEMM:  C[M,N] = A[M,K] @ W[N,K]^T + bias  (optional exact GELU)
// Block: 256 threads = 8 waves, C tile 128x128, K depth 64 (bf16 in LDS).
// Wave (wr 0..1, wc 0..3) owns 64x32 = 4x2 grid of 16x16 WMMA tiles.
// M,N multiples of 128; K multiple of 64 (true for all shapes here).
// ---------------------------------------------------------------------------
#define BM 128
#define BN 128
#define BK 64
#define LDK (BK + 8)   // 72 bf16 = 144 B row stride (16B aligned, bank-spread)

__global__ __launch_bounds__(256) void gemm_bias_act_kernel(
    const float* __restrict__ A, const float* __restrict__ W,
    const float* __restrict__ bias, float* __restrict__ C,
    int M, int N, int K, int act)
{
  __shared__ __bf16 As[BM * LDK];
  __shared__ __bf16 Bs[BN * LDK];
  const int tid  = threadIdx.x;
  const int lane = tid & 31;
  const int wave = tid >> 5;
  const int wr   = wave >> 2;
  const int wc   = wave & 3;
  const int bm   = blockIdx.y * BM;
  const int bn   = blockIdx.x * BN;
  const int r    = lane & 15;
  const int kb   = (lane >> 4) << 3;

  v8f acc[4][2] = {};

  for (int k0 = 0; k0 < K; k0 += BK) {
    __syncthreads();
#pragma unroll
    for (int it = 0; it < 8; ++it) {          // 2048 float4 chunks per tile
      int i   = tid + it * 256;
      int row = i >> 4;                       // BK/4 = 16 chunks per row
      int col = (i & 15) << 2;
      float4 f = *(const float4*)(A + (size_t)(bm + row) * K + k0 + col);
      __bf16* d = &As[row * LDK + col];
      d[0] = (__bf16)f.x; d[1] = (__bf16)f.y; d[2] = (__bf16)f.z; d[3] = (__bf16)f.w;
      float4 g = *(const float4*)(W + (size_t)(bn + row) * K + k0 + col);
      __bf16* e = &Bs[row * LDK + col];
      e[0] = (__bf16)g.x; e[1] = (__bf16)g.y; e[2] = (__bf16)g.z; e[3] = (__bf16)g.w;
    }
    __syncthreads();
#pragma unroll
    for (int ks = 0; ks < BK; ks += 32) {
      v16bf af[4], bf[2];
#pragma unroll
      for (int i = 0; i < 4; ++i)
        af[i] = load_frag16(&As[(wr * 64 + i * 16 + r) * LDK + ks + kb]);
#pragma unroll
      for (int j = 0; j < 2; ++j)
        bf[j] = load_frag16(&Bs[(wc * 32 + j * 16 + r) * LDK + ks + kb]);
#pragma unroll
      for (int i = 0; i < 4; ++i)
#pragma unroll
        for (int j = 0; j < 2; ++j)
          acc[i][j] = wmma_bf16(af[i], bf[j], acc[i][j]);
    }
  }

  // C/D layout: VGPR rr -> row rr + 8*(lane>>4), col = lane&15
  const int ch = lane & 15;
  const int rh = (lane >> 4) << 3;
#pragma unroll
  for (int i = 0; i < 4; ++i) {
#pragma unroll
    for (int j = 0; j < 2; ++j) {
      const int gcol = bn + wc * 32 + j * 16 + ch;
      const float bv = bias[gcol];
#pragma unroll
      for (int rr = 0; rr < 8; ++rr) {
        const int grow = bm + wr * 64 + i * 16 + rh + rr;
        float v = acc[i][j][rr] + bv;
        if (act == 1) v = gelu_exact(v);
        C[(size_t)grow * N + gcol] = v;
      }
    }
  }
}

// ---------------------------------------------------------------------------
// Flash attention, one (b, h, 128 q-rows) per block; wave = 16 q-rows.
// Q,K,V,O are [B*S, D_MODEL] fp32; head slice = cols h*64..h*64+63.
// ---------------------------------------------------------------------------
#define LKV (DH + 8)   // 72

__global__ __launch_bounds__(256) void attn_kernel(
    const float* __restrict__ Q, const float* __restrict__ Kb,
    const float* __restrict__ Vb, float* __restrict__ O,
    int Sq, int Skv)
{
  __shared__ __bf16 Ks[64 * LKV];
  __shared__ __bf16 Vs[64 * LKV];
  __shared__ __bf16 Ps[8 * 16 * LKV];   // per-wave 16x64 P staging

  const int tid = threadIdx.x, lane = tid & 31, wave = tid >> 5;
  const int bh = blockIdx.y;
  const int b = bh >> 4, h = bh & 15;
  const int q0 = blockIdx.x * 128 + wave * 16;
  const int r  = lane & 15;
  const int kb = (lane >> 4) << 3;
  const int rh = (lane >> 4) << 3;

  // Q fragments, pre-scaled by 1/sqrt(dh) = 0.125
  v16bf qf[2];
  {
    const float* qrow = Q + (size_t)(b * Sq + q0 + r) * D_MODEL + h * DH;
#pragma unroll
    for (int ks = 0; ks < 2; ++ks) {
      v16bf t;
#pragma unroll
      for (int j = 0; j < 8; ++j) t[j]     = (__bf16)(qrow[ks * 32 + kb + j] * 0.125f);
#pragma unroll
      for (int j = 0; j < 8; ++j) t[8 + j] = (__bf16)(qrow[ks * 32 + kb + 16 + j] * 0.125f);
      qf[ks] = t;
    }
  }

  float m[8], l[8];
#pragma unroll
  for (int i = 0; i < 8; ++i) { m[i] = -3.0e38f; l[i] = 0.0f; }
  v8f oacc[4] = {};

  __bf16* pbase = &Ps[wave * 16 * LKV];

  for (int kv0 = 0; kv0 < Skv; kv0 += 64) {
    __syncthreads();
#pragma unroll
    for (int it = 0; it < 4; ++it) {          // 64x64 fp32 -> bf16, K and V
      int i   = tid + it * 256;
      int row = i >> 4;
      int col = (i & 15) << 2;
      size_t g = (size_t)(b * Skv + kv0 + row) * D_MODEL + h * DH + col;
      float4 fk = *(const float4*)(Kb + g);
      __bf16* dk = &Ks[row * LKV + col];
      dk[0]=(__bf16)fk.x; dk[1]=(__bf16)fk.y; dk[2]=(__bf16)fk.z; dk[3]=(__bf16)fk.w;
      float4 fv = *(const float4*)(Vb + g);
      __bf16* dv = &Vs[row * LKV + col];
      dv[0]=(__bf16)fv.x; dv[1]=(__bf16)fv.y; dv[2]=(__bf16)fv.z; dv[3]=(__bf16)fv.w;
    }
    __syncthreads();

    // S = Q @ K^T   (4 tiles of 16 keys, K-dim = dh = 64 -> 2 WMMA each)
    v8f sc[4];
#pragma unroll
    for (int t = 0; t < 4; ++t) {
      v8f c = {};
#pragma unroll
      for (int ks = 0; ks < 2; ++ks) {
        v16bf bf = load_frag16(&Ks[(t * 16 + r) * LKV + ks * 32 + kb]);
        c = wmma_bf16(qf[ks], bf, c);
      }
      sc[t] = c;
    }

    // online softmax: row max/sum across 16 lanes of each half (row-major C)
    float fac[8];
#pragma unroll
    for (int rr = 0; rr < 8; ++rr) {
      float mx = fmaxf(fmaxf(sc[0][rr], sc[1][rr]), fmaxf(sc[2][rr], sc[3][rr]));
#pragma unroll
      for (int off = 8; off >= 1; off >>= 1)
        mx = fmaxf(mx, __shfl_xor(mx, off, 16));
      float mn = fmaxf(m[rr], mx);
      fac[rr] = __expf(m[rr] - mn);
      m[rr] = mn;
    }
#pragma unroll
    for (int t = 0; t < 4; ++t)
#pragma unroll
      for (int rr = 0; rr < 8; ++rr)
        sc[t][rr] = __expf(sc[t][rr] - m[rr]);
#pragma unroll
    for (int rr = 0; rr < 8; ++rr) {
      float s = sc[0][rr] + sc[1][rr] + sc[2][rr] + sc[3][rr];
#pragma unroll
      for (int off = 8; off >= 1; off >>= 1)
        s += __shfl_xor(s, off, 16);
      l[rr] = l[rr] * fac[rr] + s;
    }
#pragma unroll
    for (int t = 0; t < 4; ++t)
#pragma unroll
      for (int rr = 0; rr < 8; ++rr)
        oacc[t][rr] *= fac[rr];

    // stage P (C layout -> row-major bf16) in this wave's LDS slice
#pragma unroll
    for (int t = 0; t < 4; ++t)
#pragma unroll
      for (int rr = 0; rr < 8; ++rr)
        pbase[(rr + rh) * LKV + t * 16 + (lane & 15)] = (__bf16)sc[t][rr];

    asm volatile("s_wait_dscnt 0" ::: "memory");   // store->load, same wave

    // O += P @ V   (4 dh tiles, kv = 64 -> 2 WMMA each)
#pragma unroll
    for (int t = 0; t < 4; ++t) {
      const int dcol = t * 16 + (lane & 15);
#pragma unroll
      for (int ks = 0; ks < 2; ++ks) {
        v16bf pf = load_frag16(&pbase[r * LKV + ks * 32 + kb]);
        v16bf vf;
#pragma unroll
        for (int j = 0; j < 8; ++j) vf[j]     = Vs[(ks * 32 + kb + j)      * LKV + dcol];
#pragma unroll
        for (int j = 0; j < 8; ++j) vf[8 + j] = Vs[(ks * 32 + kb + 16 + j) * LKV + dcol];
        oacc[t] = wmma_bf16(pf, vf, oacc[t]);
      }
    }
  }

#pragma unroll
  for (int t = 0; t < 4; ++t) {
    const int gcol = h * DH + t * 16 + (lane & 15);
#pragma unroll
    for (int rr = 0; rr < 8; ++rr) {
      const int grow = b * Sq + q0 + rh + rr;
      O[(size_t)grow * D_MODEL + gcol] = oacc[t][rr] / l[rr];
    }
  }
}

// ---------------------------------------------------------------------------
__global__ __launch_bounds__(256) void add_kernel(
    const float4* __restrict__ a, const float4* __restrict__ b,
    float4* __restrict__ o)
{
  int i = blockIdx.x * 256 + threadIdx.x;
  float4 x = a[i], y = b[i];
  o[i] = make_float4(x.x + y.x, x.y + y.y, x.z + y.z, x.w + y.w);
}

// out = LayerNorm(X + Dx) * g + b  over D_MODEL; one block (256 thr) per row.
__global__ __launch_bounds__(256) void ln_residual_kernel(
    const float* __restrict__ X, const float* __restrict__ Dx,
    const float* __restrict__ g, const float* __restrict__ b,
    float* __restrict__ out)
{
  const int row = blockIdx.x;
  const float* xr = X  + (size_t)row * D_MODEL;
  const float* dr = Dx + (size_t)row * D_MODEL;
  float vals[4];
  float s = 0.f, ss = 0.f;
#pragma unroll
  for (int i = 0; i < 4; ++i) {
    int c = threadIdx.x + i * 256;
    float v = xr[c] + dr[c];
    vals[i] = v; s += v; ss += v * v;
  }
#pragma unroll
  for (int off = 16; off >= 1; off >>= 1) {
    s  += __shfl_xor(s,  off, 32);
    ss += __shfl_xor(ss, off, 32);
  }
  __shared__ float rs[8], rss[8];
  int wave = threadIdx.x >> 5;
  if ((threadIdx.x & 31) == 0) { rs[wave] = s; rss[wave] = ss; }
  __syncthreads();
  s = 0.f; ss = 0.f;
#pragma unroll
  for (int w = 0; w < 8; ++w) { s += rs[w]; ss += rss[w]; }
  const float mean = s * (1.0f / D_MODEL);
  const float var  = ss * (1.0f / D_MODEL) - mean * mean;
  const float inv  = rsqrtf(var + 1e-5f);
#pragma unroll
  for (int i = 0; i < 4; ++i) {
    int c = threadIdx.x + i * 256;
    out[(size_t)row * D_MODEL + c] = (vals[i] - mean) * inv * g[c] + b[c];
  }
}

// ---------------------------------------------------------------------------
extern "C" void kernel_launch(void* const* d_in, const int* in_sizes, int n_in,
                              void* d_out, int out_size, void* d_ws, size_t ws_size,
                              hipStream_t stream)
{
  (void)in_sizes; (void)n_in; (void)out_size; (void)ws_size;
  const float* src      = (const float*)d_in[0];
  const float* pos      = (const float*)d_in[1];
  const float* mem      = (const float*)d_in[2];
  const float* sa_w_in  = (const float*)d_in[3];
  const float* sa_b_in  = (const float*)d_in[4];
  const float* sa_w_out = (const float*)d_in[5];
  const float* sa_b_out = (const float*)d_in[6];
  const float* ca_w_in  = (const float*)d_in[7];
  const float* ca_b_in  = (const float*)d_in[8];
  const float* ca_w_out = (const float*)d_in[9];
  const float* ca_b_out = (const float*)d_in[10];
  const float* ln1_g = (const float*)d_in[11];
  const float* ln1_b = (const float*)d_in[12];
  const float* lnm_g = (const float*)d_in[13];
  const float* lnm_b = (const float*)d_in[14];
  const float* ln2_g = (const float*)d_in[15];
  const float* ln2_b = (const float*)d_in[16];
  const float* w1 = (const float*)d_in[17];
  const float* b1 = (const float*)d_in[18];
  const float* w2 = (const float*)d_in[19];
  const float* b2 = (const float*)d_in[20];
  float* out = (float*)d_out;

  const size_t TOK = (size_t)4096 * 1024;     // B*S rows x D cols
  float* ws   = (float*)d_ws;
  float* bqk  = ws;                 // src+pos; reused as attention output
  float* bq   = ws + 1 * TOK;
  float* bk   = ws + 2 * TOK;
  float* bv   = ws + 3 * TOK;
  float* btmp = ws + 4 * TOK;
  float* bx1  = ws + 5 * TOK;
  float* bx2  = ws + 6 * TOK;
  float* bh   = ws + 7 * TOK;       // 4096 x 4096

  dim3 blk(256);
  dim3 gAdd((unsigned)(TOK / (256 * 4)));
  dim3 gGemmD(1024 / 128, 4096 / 128);    // N=1024
  dim3 gGemmF1(4096 / 128, 4096 / 128);   // N=4096
  dim3 gAttn(1024 / 128, 4 * 16);
  dim3 gLN(4096);

  // --- self attention (q,k from src+pos, v from src) ---
  add_kernel<<<gAdd, blk, 0, stream>>>((const float4*)src, (const float4*)pos, (float4*)bqk);
  gemm_bias_act_kernel<<<gGemmD, blk, 0, stream>>>(bqk, sa_w_in,               sa_b_in,        bq, 4096, 1024, 1024, 0);
  gemm_bias_act_kernel<<<gGemmD, blk, 0, stream>>>(bqk, sa_w_in + 1024*1024,   sa_b_in + 1024, bk, 4096, 1024, 1024, 0);
  gemm_bias_act_kernel<<<gGemmD, blk, 0, stream>>>(src, sa_w_in + 2*1024*1024, sa_b_in + 2048, bv, 4096, 1024, 1024, 0);
  attn_kernel<<<gAttn, blk, 0, stream>>>(bq, bk, bv, bqk, 1024, 1024);
  gemm_bias_act_kernel<<<gGemmD, blk, 0, stream>>>(bqk, sa_w_out, sa_b_out, btmp, 4096, 1024, 1024, 0);
  ln_residual_kernel<<<gLN, blk, 0, stream>>>(src, btmp, ln1_g, ln1_b, bx1);

  // --- cross attention (q from x, k,v from mem) ---
  gemm_bias_act_kernel<<<gGemmD, blk, 0, stream>>>(bx1, ca_w_in,               ca_b_in,        bq, 4096, 1024, 1024, 0);
  gemm_bias_act_kernel<<<gGemmD, blk, 0, stream>>>(mem, ca_w_in + 1024*1024,   ca_b_in + 1024, bk, 4096, 1024, 1024, 0);
  gemm_bias_act_kernel<<<gGemmD, blk, 0, stream>>>(mem, ca_w_in + 2*1024*1024, ca_b_in + 2048, bv, 4096, 1024, 1024, 0);
  attn_kernel<<<gAttn, blk, 0, stream>>>(bq, bk, bv, bqk, 1024, 1024);
  gemm_bias_act_kernel<<<gGemmD, blk, 0, stream>>>(bqk, ca_w_out, ca_b_out, btmp, 4096, 1024, 1024, 0);
  ln_residual_kernel<<<gLN, blk, 0, stream>>>(bx1, btmp, lnm_g, lnm_b, bx2);

  // --- FFN with exact GELU ---
  gemm_bias_act_kernel<<<gGemmF1, blk, 0, stream>>>(bx2, w1, b1, bh,   4096, 4096, 1024, 1);
  gemm_bias_act_kernel<<<gGemmD,  blk, 0, stream>>>(bh,  w2, b2, btmp, 4096, 1024, 4096, 0);
  ln_residual_kernel<<<gLN, blk, 0, stream>>>(bx2, btmp, ln2_g, ln2_b, out);
}
